// GCN_6390911336843
// MI455X (gfx1250) — compile-verified, matrix operand order checked by reading, not run on hardware
//
#include <hip/hip_runtime.h>
#include <hip/hip_bf16.h>

typedef __attribute__((ext_vector_type(2)))  float    v2f;
typedef __attribute__((ext_vector_type(8)))  float    v8f;
typedef __attribute__((ext_vector_type(16))) _Float16 v16h;

#define CH 64  // IN_CH == HID == 64

// ---------------------------------------------------------------------------
// Degree / normalization kernels
// ---------------------------------------------------------------------------
__global__ void k_fill(float* __restrict__ p, float v, int n) {
  int i = blockIdx.x * blockDim.x + threadIdx.x;
  if (i < n) p[i] = v;
}

__global__ void k_deg(const int* __restrict__ dst, float* __restrict__ deg, int E) {
  int e = blockIdx.x * blockDim.x + threadIdx.x;
  if (e < E) unsafeAtomicAdd(&deg[dst[e]], 1.0f);  // global_atomic_add_f32
}

__global__ void k_rsqrt(float* __restrict__ p, int n) {
  int i = blockIdx.x * blockDim.x + threadIdx.x;
  if (i < n) p[i] = rsqrtf(p[i]);  // deg >= 1 always (self-loops)
}

__global__ void k_edge_norm(const int* __restrict__ src, const int* __restrict__ dst,
                            const float* __restrict__ dinv, float* __restrict__ nrm, int E) {
  int e = blockIdx.x * blockDim.x + threadIdx.x;
  if (e < E) nrm[e] = dinv[src[e]] * dinv[dst[e]];
}

// ---------------------------------------------------------------------------
// Dense linear:  C[N,64] = A[N,64] @ W[64,64]   via CDNA5 WMMA
// Block = 256 threads = 8 waves; wave (rt,ct) computes a 16x16 tile:
//   rows  blockIdx.x*32 + rt*16 .. +15,  cols ct*16 .. +15
// Fragment layout (ISA 7.12.2, wave32):
//   A 16x4 f32 : lane = {kh = lane>>4 selects K pair, m = lane&15 = row};
//                v2f = a[m][k0+2kh], a[m][k0+2kh+1]
//   B 4x16 f32 : lane = {kh, n = lane&15};  v2f = b[k0+2kh][n], b[k0+2kh+1][n]
//   C 16x16 f32: VGPR v, lanes 0-15 -> M=v, lanes 16-31 -> M=v+8; N = lane&15
// ---------------------------------------------------------------------------
__global__ __launch_bounds__(256) void k_gemm64(const float* __restrict__ A,
                                                const float* __restrict__ W,
                                                float* __restrict__ C, int N) {
  const int lane = threadIdx.x & 31;
  const int wave = threadIdx.x >> 5;
  const int row0 = blockIdx.x * 32 + (wave >> 2) * 16;
  if (row0 >= N) return;                 // wave-uniform: EXEC stays all-ones for WMMA
  const int n0 = (wave & 3) * 16;
  const int m  = lane & 15;              // A row / B,C col within tile
  const int kh = lane >> 4;              // K-pair (A/B) and row-half (C) select
  v8f acc = {};

#if __has_builtin(__builtin_amdgcn_wmma_f32_16x16x4_f32)
  const float* arow = A + (size_t)(row0 + m) * CH + 2 * kh;
#pragma unroll
  for (int k0 = 0; k0 < CH; k0 += 4) {
    v2f a;
    a.x = arow[k0];
    a.y = arow[k0 + 1];
    v2f b;
    b.x = W[(k0 + 2 * kh + 0) * CH + n0 + m];
    b.y = W[(k0 + 2 * kh + 1) * CH + n0 + m];
    acc = __builtin_amdgcn_wmma_f32_16x16x4_f32(false, a, false, b,
                                                (short)0, acc, false, false);
  }
#else
  // Fallback: codegen-confirmed f16 WMMA, two K-blocks of 32, convert on the fly.
  // 16-bit A fragment: element e<8 -> K = 8*kh+e ; e>=8 -> K = 16+8*kh+(e-8)
  const float* arow = A + (size_t)(row0 + m) * CH;
#pragma unroll
  for (int kb = 0; kb < CH; kb += 32) {
    v16h a, b;
#pragma unroll
    for (int e = 0; e < 16; ++e) {
      int k = (e < 8) ? (8 * kh + e) : (16 + 8 * kh + (e - 8));
      a[e] = (_Float16)arow[kb + k];
      b[e] = (_Float16)W[(size_t)(kb + k) * CH + n0 + m];
    }
    acc = __builtin_amdgcn_wmma_f32_16x16x32_f16(false, a, false, b,
                                                 (short)0, acc, false, false);
  }
#endif

#pragma unroll
  for (int v = 0; v < 8; ++v) {
    int r = row0 + v + 8 * kh;
    if (r < N) C[(size_t)r * CH + (n0 + m)] = acc[v];
  }
}

// ---------------------------------------------------------------------------
// out[i,:] = h[i,:] * dinv[i]^2   (self-loop term; also serves as the zero-init)
// ---------------------------------------------------------------------------
__global__ void k_selfloop(const float* __restrict__ h, const float* __restrict__ dinv,
                           float* __restrict__ out, int total) {
  int t = blockIdx.x * blockDim.x + threadIdx.x;
  if (t >= total) return;
  float di = dinv[t >> 6];
  out[t] = h[t] * di * di;
}

// ---------------------------------------------------------------------------
// Edge scatter: wave32 per edge, float2 (2 channels) per lane, hw f32 atomics.
// Features (12.8 MB) are L2-resident on MI455X -> atomics resolve in L2.
// ---------------------------------------------------------------------------
__global__ __launch_bounds__(256) void k_scatter(const float* __restrict__ h,
                                                 const int* __restrict__ src,
                                                 const int* __restrict__ dst,
                                                 const float* __restrict__ nrm,
                                                 float* __restrict__ out, int E) {
  int t = blockIdx.x * blockDim.x + threadIdx.x;
  int e = t >> 5;
  if (e >= E) return;
  int c = (t & 31) * 2;
  int s = src[e], d = dst[e];
  float w = nrm[e];
  const float2 hv = *(const float2*)(h + (size_t)s * CH + c);
  float* op = out + (size_t)d * CH + c;
  unsafeAtomicAdd(op,     hv.x * w);
  unsafeAtomicAdd(op + 1, hv.y * w);
}

__global__ void k_bias_relu(float* __restrict__ h, const float* __restrict__ b, int total) {
  int t = blockIdx.x * blockDim.x + threadIdx.x;
  if (t >= total) return;
  float v = h[t] + b[t & (CH - 1)];
  h[t] = v > 0.0f ? v : 0.0f;
}

// ---------------------------------------------------------------------------
// Edge predictor: wave32 per edge, 4-channel dot per lane, shfl_xor reduction.
// pred[e] = concat(h[src], h[dst]) . Wfc + bfc
// ---------------------------------------------------------------------------
__global__ __launch_bounds__(256) void k_edge_pred(const float* __restrict__ h,
                                                   const int* __restrict__ src,
                                                   const int* __restrict__ dst,
                                                   const float* __restrict__ Wfc,
                                                   const float* __restrict__ bfc,
                                                   float* __restrict__ out, int E) {
  int t = blockIdx.x * blockDim.x + threadIdx.x;
  int e = t >> 5;
  if (e >= E) return;
  int lane = t & 31;
  int c = lane * 2;
  int s = src[e], d = dst[e];
  float2 hs = *(const float2*)(h + (size_t)s * CH + c);
  float2 hd = *(const float2*)(h + (size_t)d * CH + c);
  float2 w0 = *(const float2*)(Wfc + c);
  float2 w1 = *(const float2*)(Wfc + CH + c);
  float sum = hs.x * w0.x + hs.y * w0.y + hd.x * w1.x + hd.y * w1.y;
#pragma unroll
  for (int off = 16; off > 0; off >>= 1) sum += __shfl_xor(sum, off, 32);
  if (lane == 0) out[e] = sum + bfc[0];
}

// ---------------------------------------------------------------------------
// Launcher
// ---------------------------------------------------------------------------
extern "C" void kernel_launch(void* const* d_in, const int* in_sizes, int n_in,
                              void* d_out, int out_size, void* d_ws, size_t ws_size,
                              hipStream_t stream) {
  const float* x   = (const float*)d_in[0];
  const int*   ei  = (const int*)d_in[1];
  const float* W1  = (const float*)d_in[2];
  const float* b1  = (const float*)d_in[3];
  const float* W2  = (const float*)d_in[4];
  const float* b2  = (const float*)d_in[5];
  const float* Wfc = (const float*)d_in[6];
  const float* bfc = (const float*)d_in[7];

  const int N = in_sizes[0] / CH;      // 50000
  const int E = in_sizes[1] / 2;       // 800000
  const int NF = N * CH;               // node-feature element count

  const int* src = ei;
  const int* dst = ei + E;

  // Workspace layout (floats): dinv[N] | norm[E] | HBUF[N*64] | ABUF[N*64]
  float* ws   = (float*)d_ws;
  float* dinv = ws;
  float* nrm  = dinv + N;
  float* HB   = nrm + E;
  float* AB   = HB + (size_t)NF;

  const int TB = 256;
  dim3 blk(TB);
  dim3 gN((N + TB - 1) / TB);
  dim3 gE((E + TB - 1) / TB);
  dim3 gNF((NF + TB - 1) / TB);
  dim3 gEw((E * 32 + TB - 1) / TB);    // wave-per-edge kernels
  dim3 gG((N + 31) / 32);              // GEMM: 32 rows / block

  // --- normalization (shared by both layers) ---
  k_fill<<<gN, blk, 0, stream>>>(dinv, 1.0f, N);                    // self-loop degree
  k_deg<<<gE, blk, 0, stream>>>(dst, dinv, E);
  k_rsqrt<<<gN, blk, 0, stream>>>(dinv, N);
  k_edge_norm<<<gE, blk, 0, stream>>>(src, dst, dinv, nrm, E);

  // --- layer 1 ---
  k_gemm64<<<gG, blk, 0, stream>>>(x, W1, HB, N);
  k_selfloop<<<gNF, blk, 0, stream>>>(HB, dinv, AB, NF);
  k_scatter<<<gEw, blk, 0, stream>>>(HB, src, dst, nrm, AB, E);
  k_bias_relu<<<gNF, blk, 0, stream>>>(AB, b1, NF);

  // --- layer 2 (reuse HBUF/ABUF) ---
  k_gemm64<<<gG, blk, 0, stream>>>(AB, W2, HB, N);
  k_selfloop<<<gNF, blk, 0, stream>>>(HB, dinv, AB, NF);
  k_scatter<<<gEw, blk, 0, stream>>>(HB, src, dst, nrm, AB, E);
  k_bias_relu<<<gNF, blk, 0, stream>>>(AB, b2, NF);

  // --- edge MLP head ---
  k_edge_pred<<<gEw, blk, 0, stream>>>(AB, src, dst, Wfc, bfc, (float*)d_out, E);
}
